// Get_adaptivesmoothtar_36146444763361
// MI455X (gfx1250) — matrix operand order repaired and checked by reading, not compiled, not run on hardware
//
#include <hip/hip_runtime.h>

// ---- CDNA5 wave32 WMMA types -------------------------------------------------
typedef __attribute__((ext_vector_type(16))) _Float16 v16h;
typedef __attribute__((ext_vector_type(8)))  float    v8f;
typedef __attribute__((ext_vector_type(8)))  int      v8i;

union Frag { v8i i; v16h h; };

#define IMG_H 512
#define IMG_W 512
#define NCH   3
#define NBATCH 16
#define TILE  16
#define TILES_X (IMG_W / TILE)            // 32
#define TILES_Y (IMG_H / TILE)            // 32
#define NTILES (NBATCH * TILES_X * TILES_Y) // 16384
#define WAVES_PER_BLOCK 4

// f32 stage tile: 32 rows (K dim, rows 22..31 kept zero) x 32 cols, stride 33
// floats; lanes 0-15 read rows k and lanes 16-31 rows k+8.
#define ST_STRIDE 33
#define ST_ROWS   32
#define ST_SIZE   (ST_ROWS * ST_STRIDE)   // 1056 floats
// T relayout tiles (f16), row-major [16][34] so K-pairs are aligned b32.
#define T_STRIDE  34
#define T_SIZE    (16 * T_STRIDE)         // 544 halfs

// Normalized 1D Gaussian (size 5, sigma 1): g = {G0,G1,G2,G1,G0}
#define G0 0.054488685f
#define G1 0.244201342f
#define G2 0.402619958f

// Composite 7-tap kernels: full conv of gauss with [1,2,1] (smooth) / [-1,0,1] (diff)
__device__ __forceinline__ float band_sm(int d) {
    switch (d) {
        case 0: case 6: return G0;
        case 1: case 5: return 2.f * G0 + G1;
        case 2: case 4: return G0 + 2.f * G1 + G2;
        case 3:         return 2.f * G1 + 2.f * G2;
        default:        return 0.f;
    }
}
__device__ __forceinline__ float band_df(int d) {
    switch (d) {
        case 0: return -G0;
        case 1: return -G1;
        case 2: return G0 - G2;
        case 4: return G2 - G0;
        case 5: return G1;
        case 6: return G0;
        default: return 0.f;
    }
}

__device__ __forceinline__ v8f wmma_f16(v16h a, v16h b, v8f c) {
    return __builtin_amdgcn_wmma_f32_16x16x32_f16(false, a, false, b, (short)0, c,
                                                  false, false);
}

// ---- CDNA5 async global->LDS staging (guarded) -------------------------------
#if __has_builtin(__builtin_amdgcn_global_load_async_to_lds_b32) && \
    __has_builtin(__builtin_amdgcn_s_wait_asynccnt)
#define HAVE_ASYNC_LDS 1
#else
#define HAVE_ASYNC_LDS 0
#endif

#if HAVE_ASYNC_LDS
typedef __attribute__((address_space(1))) int* gptr_t;
typedef __attribute__((address_space(3))) int* lptr_t;
__device__ __forceinline__ void async_ld_b32(const float* g, float* l) {
    __builtin_amdgcn_global_load_async_to_lds_b32(
        (gptr_t)(uintptr_t)g,
        (lptr_t)(unsigned int)(uintptr_t)l,
        0, 0);
}
#endif

// Kernel 1: per 16x16 tile, sum-over-channels gradient magnitude via WMMA.
__global__ __launch_bounds__(128) void canny_mag_kernel(const float* __restrict__ in,
                                                        float* __restrict__ mag) {
    __shared__ float    s_stage[WAVES_PER_BLOCK][2][ST_SIZE];
    __shared__ _Float16 s_ts[WAVES_PER_BLOCK][T_SIZE];
    __shared__ _Float16 s_td[WAVES_PER_BLOCK][T_SIZE];

    const int wave = threadIdx.x >> 5;
    const int lane = threadIdx.x & 31;
    const int tile = blockIdx.x * WAVES_PER_BLOCK + wave;   // grid sized exactly

    const int b  = tile >> 10;
    const int ty = (tile >> 5) & 31;
    const int tx = tile & 31;
    const int y0 = ty * TILE;
    const int x0 = tx * TILE;
    const bool interior = (tx >= 1) && (tx <= 30) && (ty >= 1) && (ty <= 30);

    const int nIdx = lane & 15;          // N (or M for A-frags)
    const int hi   = (lane >> 4) & 1;    // lane half

    float*     stg[2] = { s_stage[wave][0], s_stage[wave][1] };
    _Float16*  ts = s_ts[wave];
    _Float16*  td = s_td[wave];

    // Constant band fragments (per the Sec 7.12.2 f16 layouts).
    // A (16x32): M = lane&15; slot v, pair e: K = 2*(v&3) + 16*(v>>2) + 8*hi + e
    // B (32x16): N = lane&15; slot v, pair e: K = 2*(v&3) + 16*(v>>2) + 8*hi + e
    v16h a_sm, a_df, b_sm, b_df;
#pragma unroll
    for (int v = 0; v < 8; ++v) {
#pragma unroll
        for (int e = 0; e < 2; ++e) {
            const int idx = 2 * v + e;
            const int dA = 2 * (v & 3) + ((v >> 2) << 4) + (hi << 3) + e - nIdx;
            a_sm[idx] = (_Float16)band_sm(dA);
            a_df[idx] = (_Float16)band_df(dA);
            const int dB = 2 * (v & 3) + ((v >> 2) << 4) + (hi << 3) + e - nIdx;
            b_sm[idx] = (_Float16)band_sm(dB);
            b_df[idx] = (_Float16)band_df(dB);
        }
    }

    // Zero stage rows 22..31 (K-padding rows: must stay finite-zero).
    for (int i = lane; i < (ST_ROWS - 22) * ST_STRIDE; i += 32) {
        stg[0][22 * ST_STRIDE + i] = 0.f;
        stg[1][22 * ST_STRIDE + i] = 0.f;
    }
    asm volatile("" ::: "memory");

    // Stage rows 0..21 = input rows y0-3 .. y0+18; cols = x0-3+lane.
    auto load_channel = [&](int c, float* buf) {
        const float* __restrict__ src =
            in + ((size_t)b * NCH + c) * (size_t)(IMG_H * IMG_W);
        const int gy = y0 - 3, gc0 = x0 - 3;
#if HAVE_ASYNC_LDS
        if (interior) {   // wave-uniform; fully unchecked, async, double-buffered
            const float* g = src + (size_t)gy * IMG_W + gc0 + lane;
            float* l = buf + lane;
#pragma unroll
            for (int r = 0; r < 22; ++r) {
                async_ld_b32(g, l);
                g += IMG_W;
                l += ST_STRIDE;
            }
            return;
        }
#else
        if (interior) {   // unchecked synchronous path
            const float* g = src + (size_t)gy * IMG_W + gc0 + lane;
#pragma unroll
            for (int r = 0; r < 22; ++r)
                buf[r * ST_STRIDE + lane] = g[(size_t)r * IMG_W];
            return;
        }
#endif
        // Border tiles: zero-padded loads.
        for (int r = 0; r < 22; ++r) {
            const int gr = gy + r;
            const int gc = gc0 + lane;
            float v = 0.f;
            if (gr >= 0 && gr < IMG_H && gc >= 0 && gc < IMG_W)
                v = src[gr * IMG_W + gc];
            buf[r * ST_STRIDE + lane] = v;
        }
    };

    load_channel(0, stg[0]);

    v8f macc = {};
    const v8f zero = {};

    for (int c = 0; c < NCH; ++c) {
        if (c + 1 < NCH) load_channel(c + 1, stg[(c + 1) & 1]);
#if HAVE_ASYNC_LDS
        // Async loads complete in order: <=22 outstanding means channel c landed.
        if (c < NCH - 1) __builtin_amdgcn_s_wait_asynccnt(22);
        else             __builtin_amdgcn_s_wait_asynccnt(0);
#endif
        asm volatile("" ::: "memory");
        const float* st = stg[c & 1];

        // Stage-1 B frags: B[K=row][N=col(16h+n)] from f32 stage, cvt in regs.
        Frag bx0, bx1;
#pragma unroll
        for (int v = 0; v < 8; ++v) {
            const int k = 2 * (v & 3) + ((v >> 2) << 4) + (hi << 3);
            bx0.h[2 * v]     = (_Float16)st[k * ST_STRIDE + nIdx];
            bx0.h[2 * v + 1] = (_Float16)st[(k + 1) * ST_STRIDE + nIdx];
            bx1.h[2 * v]     = (_Float16)st[k * ST_STRIDE + 16 + nIdx];
            bx1.h[2 * v + 1] = (_Float16)st[(k + 1) * ST_STRIDE + 16 + nIdx];
        }

        // Stage 1: vertical composite convs (smooth & diff), two column halves.
        v8f ts0 = wmma_f16(a_sm, bx0.h, zero);
        v8f ts1 = wmma_f16(a_sm, bx1.h, zero);
        v8f td0 = wmma_f16(a_df, bx0.h, zero);
        v8f td1 = wmma_f16(a_df, bx1.h, zero);

        // Relayout T (D-frag f32) -> LDS f16 row-major [M][K].
#pragma unroll
        for (int v = 0; v < 8; ++v) {
            const int row = v + (hi << 3);
            ts[row * T_STRIDE + nIdx]      = (_Float16)ts0[v];
            ts[row * T_STRIDE + 16 + nIdx] = (_Float16)ts1[v];
            td[row * T_STRIDE + nIdx]      = (_Float16)td0[v];
            td[row * T_STRIDE + 16 + nIdx] = (_Float16)td1[v];
        }
        asm volatile("" ::: "memory");

        // Stage-2 A frags from LDS.
        Frag ats, atd;
#pragma unroll
        for (int v = 0; v < 8; ++v) {
            const int k = 2 * (v & 3) + ((v >> 2) << 4) + (hi << 3);
            ats.i[v] = *(const int*)(ts + nIdx * T_STRIDE + k);
            atd.i[v] = *(const int*)(td + nIdx * T_STRIDE + k);
        }

        // Stage 2: horizontal composite convs -> gx, gy (16x16, C/D layout).
        v8f gx = wmma_f16(ats.h, b_df, zero);
        v8f gy = wmma_f16(atd.h, b_sm, zero);

#pragma unroll
        for (int v = 0; v < 8; ++v)
            macc[v] += sqrtf(gx[v] * gx[v] + gy[v] * gy[v] + 1e-12f);
    }

    // Store magnitude tile (C/D layout: row = v + 8*hi, col = nIdx).
    float* __restrict__ mout = mag + (size_t)b * (IMG_H * IMG_W);
#pragma unroll
    for (int v = 0; v < 8; ++v)
        mout[(y0 + v + (hi << 3)) * IMG_W + x0 + nIdx] = macc[v];
}

// Kernel 2: hysteresis threshold + scale + clamp. mag (16.8MB) is L2-resident.
__global__ __launch_bounds__(256) void hysteresis_kernel(const float* __restrict__ mag,
                                                         float* __restrict__ out) {
    const int idx = blockIdx.x * 256 + threadIdx.x;      // 16*512*512 exactly
    const int b = idx >> 18;
    const int y = (idx >> 9) & 511;
    const int x = idx & 511;
    const float* __restrict__ m = mag + ((size_t)b << 18);

    const float mv = m[(y << 9) | x];

    // clamped 3x3 max == zero-padded dilation (values are >= 0)
    const int ym = (y > 0)   ? y - 1 : 0;
    const int yp = (y < 511) ? y + 1 : 511;
    const int xm = (x > 0)   ? x - 1 : 0;
    const int xp = (x < 511) ? x + 1 : 511;

    float smax = m[(ym << 9) | xm];
    smax = fmaxf(smax, m[(ym << 9) | x ]);
    smax = fmaxf(smax, m[(ym << 9) | xp]);
    smax = fmaxf(smax, m[(y  << 9) | xm]);
    smax = fmaxf(smax, mv);
    smax = fmaxf(smax, m[(y  << 9) | xp]);
    smax = fmaxf(smax, m[(yp << 9) | xm]);
    smax = fmaxf(smax, m[(yp << 9) | x ]);
    smax = fmaxf(smax, m[(yp << 9) | xp]);

    const bool keep = (mv > 0.1f) && (smax > 0.5f);
    const float scaled = fminf(mv * (1.5f / 16.971f), 1.0f);
    out[idx] = keep ? scaled : 0.f;
}

extern "C" void kernel_launch(void* const* d_in, const int* in_sizes, int n_in,
                              void* d_out, int out_size, void* d_ws, size_t ws_size,
                              hipStream_t stream) {
    (void)in_sizes; (void)n_in; (void)out_size; (void)ws_size;
    const float* in = (const float*)d_in[0];
    float* out = (float*)d_out;
    float* mag = (float*)d_ws;   // needs 16*512*512*4 = 16 MiB scratch

    canny_mag_kernel<<<NTILES / WAVES_PER_BLOCK, 128, 0, stream>>>(in, mag);
    hysteresis_kernel<<<(NBATCH * IMG_H * IMG_W) / 256, 256, 0, stream>>>(mag, out);
}
